// GNN_67740224193373
// MI455X (gfx1250) — compile-verified, matrix operand order checked by reading, not run on hardware
//
#include <hip/hip_runtime.h>
#include <hip/hip_bf16.h>

// MI455X (gfx1250) implementation.
// Bandwidth-bound on the 512x512x64 node-pair tensor x (kept f16, 32MB).
// Round structure: col stats -> edge features -> WMMA edge GEMM (compact m)
// -> batched GRU scan with WMMA gate GEMMs and async global->LDS staging.
// The x_new tensor is never materialized: a dedup (u,v)->edge map redirects
// GRU reads, saving ~96MB of HBM traffic per round.

typedef _Float16 f16;
typedef __attribute__((ext_vector_type(16))) _Float16 v16h;
typedef __attribute__((ext_vector_type(8)))  float    v8f;

#define VV 512
#define HH 64
#define EE 8192
#define PP 64
#define DD 68
#define G3 204   // 3*DD

__device__ __forceinline__ float sigf(float x) { return 1.0f / (1.0f + expf(-x)); }

// CDNA5 16-bit A/B fragment: lane (sub=l>>4, idx=l&15) holds row idx with
// K = sub*8 + {0..7} and sub*8 + 16 + {0..7}: two 16B chunks 32B apart.
__device__ __forceinline__ v16h load_frag16(const f16* p) {
    union { float4 f[2]; v16h v; } u;
    const float4* q = reinterpret_cast<const float4*>(p);
    u.f[0] = q[0];
    u.f[1] = q[2];
    return u.v;
}

__device__ __forceinline__ v8f wmma_f16(v16h a, v16h b, v8f c) {
    return __builtin_amdgcn_wmma_f32_16x16x32_f16(false, a, false, b, (short)0, c,
                                                  false, false);
}

// Async global->LDS (16B per lane), tracked by ASYNCcnt.
__device__ __forceinline__ void async_ld_b128(unsigned lds_addr, const f16* gaddr) {
    asm volatile("global_load_async_to_lds_b128 %0, %1, off"
                 :: "v"(lds_addr), "v"(gaddr) : "memory");
}

// ---------------- x = adj @ emb_w.T  (f32 in, f16 out) ----------------
__global__ __launch_bounds__(256) void embed_kernel(const float* __restrict__ adj,
                                                    const float* __restrict__ emb_w,
                                                    f16* __restrict__ x) {
    size_t gid = (size_t)blockIdx.x * 256 + threadIdx.x;   // V*V*H
    size_t rowp = gid >> 6;
    int h = (int)(gid & 63);
    const float* a = adj + rowp * 8;
    const float* w = emb_w + h * 8;
    float s = 0.f;
#pragma unroll
    for (int f = 0; f < 8; ++f) s += a[f] * w[f];
    x[gid] = (f16)s;
}

// ---------------- edge scatter map (built once) ----------------
__global__ void init_map_kernel(int* __restrict__ map) {
    size_t i = (size_t)blockIdx.x * 256 + threadIdx.x;     // V*V
    map[i] = -1;
}

__global__ void build_map_kernel(const int* __restrict__ edges, int* __restrict__ map) {
    int e = blockIdx.x * 256 + threadIdx.x;                // E
    int u = edges[2 * e], v = edges[2 * e + 1];
    atomicMax(&map[u * VV + v], e);   // deterministic dedup (max edge id wins)
}

// ---------------- weight conversion to f16 ----------------
__global__ void conv_wgru_kernel(const float* __restrict__ wih,
                                 const float* __restrict__ whh,
                                 f16* __restrict__ wg) {
    int id = blockIdx.x * 256 + threadIdx.x;       // 192*192
    if (id >= 192 * 192) return;
    int o = id / 192, k = id % 192;
    float v = (k < 128) ? wih[o * 128 + k] : whh[o * 64 + (k - 128)];
    wg[id] = (f16)v;
}

__global__ void conv_wmp_kernel(const float* __restrict__ w1,
                                const float* __restrict__ w2,
                                const float* __restrict__ w3,
                                f16* __restrict__ wmp) {
    int id = blockIdx.x * 256 + threadIdx.x;       // 64*192
    if (id >= 64 * 192) return;
    int o = id / 192, k = id % 192;
    float v = (k < 64) ? w1[o * 64 + k]
            : (k < 128) ? w2[o * 64 + (k - 64)]
                        : w3[o * 64 + (k - 128)];
    wmp[id] = (f16)v;
}

// ---------------- column sums + nonzero counts ----------------
__global__ __launch_bounds__(256) void col_sum_kernel(const f16* __restrict__ x,
                                                      float* __restrict__ col_sum,
                                                      float* __restrict__ col_nz) {
    __shared__ float red[256];
    __shared__ float rednz[256];
    int j = blockIdx.x;
    int t = threadIdx.x, h = t & 63, iq = t >> 6;
    float s = 0.f, nz = 0.f;
    for (int i = iq; i < VV; i += 4) {
        float v = (float)x[((size_t)i * VV + j) * HH + h];
        s += v;
        nz += (v != 0.0f) ? 1.0f : 0.0f;
    }
    red[t] = s; rednz[t] = nz;
    __syncthreads();
    if (t < 64) col_sum[j * HH + h] = red[t] + red[t + 64] + red[t + 128] + red[t + 192];
    for (int off = 128; off >= 1; off >>= 1) {
        if (t < off) rednz[t] += rednz[t + off];
        __syncthreads();
    }
    if (t == 0) col_nz[j] = rednz[0];
}

// ---------------- per-edge feature build (wave per edge) ----------------
__global__ __launch_bounds__(256) void edge_feat_kernel(const f16* __restrict__ x,
                                                        const float* __restrict__ col_sum,
                                                        const float* __restrict__ col_nz,
                                                        const int* __restrict__ edges,
                                                        f16* __restrict__ feat) {
    int wv = threadIdx.x >> 5, lane = threadIdx.x & 31;
    int e = blockIdx.x * 8 + wv;
    int u = edges[2 * e], v = edges[2 * e + 1];
    size_t base_vu = ((size_t)v * VV + u) * HH;
    size_t base_uv = ((size_t)u * VV + v) * HH;
    float xvu0 = (float)x[base_vu + lane], xvu1 = (float)x[base_vu + lane + 32];
    float xuv0 = (float)x[base_uv + lane], xuv1 = (float)x[base_uv + lane + 32];
    int nz_vu = __popcll(__ballot(xvu0 != 0.0f)) + __popcll(__ballot(xvu1 != 0.0f));
    int nz_uv = __popcll(__ballot(xuv0 != 0.0f)) + __popcll(__ballot(xuv1 != 0.0f));
    float nin = (col_nz[u] - (float)nz_vu) * (1.0f / 64.0f);
    if (nin == 0.0f) nin = 1.0f;
    float nout = (col_nz[v] - (float)nz_uv) * (1.0f / 64.0f);
    if (nout == 0.0f) nout = 1.0f;
    f16* fr = feat + (size_t)e * 192;
    fr[lane]            = (f16)xuv0;
    fr[lane + 32]       = (f16)xuv1;
    fr[64 + lane]       = (f16)((col_sum[u * HH + lane]      - xvu0) / nin);
    fr[64 + lane + 32]  = (f16)((col_sum[u * HH + lane + 32] - xvu1) / nin);
    fr[128 + lane]      = (f16)((col_sum[v * HH + lane]      - xuv0) / nout);
    fr[128 + lane + 32] = (f16)((col_sum[v * HH + lane + 32] - xuv1) / nout);
}

// ------- m = relu(feat @ [w1|w2|w3].T) -> compact E x 64 buffer (WMMA) -------
__global__ __launch_bounds__(128) void mp_gemm_kernel(const f16* __restrict__ feat,
                                                      const f16* __restrict__ wmp,
                                                      f16* __restrict__ m) {
    int tid = threadIdx.x, wave = tid >> 5, lane = tid & 31;
    int idx = lane & 15, sub = lane >> 4;
    int mt = blockIdx.x * 4 + wave;    // 512 M-tiles total
    int e0 = mt * 16;
    v16h A[6];
#pragma unroll
    for (int c = 0; c < 6; ++c)
        A[c] = load_frag16(feat + (size_t)(e0 + idx) * 192 + c * 32 + sub * 8);
#pragma unroll
    for (int nt = 0; nt < 4; ++nt) {
        v8f acc = {};
#pragma unroll
        for (int c = 0; c < 6; ++c) {
            v16h b = load_frag16(wmp + (size_t)(nt * 16 + idx) * 192 + c * 32 + sub * 8);
            acc = wmma_f16(A[c], b, acc);
        }
        int o = nt * 16 + idx;
#pragma unroll
        for (int r = 0; r < 8; ++r) {
            int e = e0 + r + sub * 8;
            float mv = acc[r] > 0.f ? acc[r] : 0.f;
            m[(size_t)e * HH + o] = (f16)mv;
        }
    }
}

// ------- GRU over node pairs: 16 batch rows per WG, 512 sequential steps -------
// inp_t = [x_new | x_old], where x_new[b,s] = m[map[b,s]] if mapped else x_old.
// Staging is double-buffered via global_load_async_to_lds_b128 (ASYNCcnt) so
// step s+1's HBM traffic overlaps step s's WMMAs.
__global__ __launch_bounds__(128) void gru_kernel(const f16* __restrict__ mbuf,
                                                  const int* __restrict__ map,
                                                  const f16* __restrict__ xold,
                                                  const f16* __restrict__ wg,
                                                  const float* __restrict__ b_ih,
                                                  const float* __restrict__ b_hh,
                                                  f16* __restrict__ xout) {
    __shared__ alignas(16) f16 Xst[2][16][128];  // double-buffered [xnew|xold]
    __shared__ alignas(16) f16 Hst[16][64];      // h as f16 (A chunks 4,5)
    __shared__ float Grz[16][128];
    __shared__ float Ginn[16][64];
    __shared__ float Ghn[16][64];
    __shared__ float Hc[16][64];
    const int tid = threadIdx.x;
    const int wave = tid >> 5, lane = tid & 31;
    const int idx = lane & 15, sub = lane >> 4;
    const int b0 = blockIdx.x * 16;
    const int t0 = wave * 3;                   // 3 output tiles per wave (12 total)

    // loop-invariant weight fragments held in VGPRs
    v16h B[3][6];
#pragma unroll
    for (int i = 0; i < 3; ++i)
#pragma unroll
        for (int c = 0; c < 6; ++c)
            B[i][c] = load_frag16(wg + (size_t)((t0 + i) * 16 + idx) * 192 + c * 32 + sub * 8);

    for (int f = tid; f < 16 * 64; f += 128) {
        Hc[f >> 6][f & 63] = 0.0f;
        Hst[f >> 6][f & 63] = (f16)0.0f;
    }

    // issue async staging for step s1 into Xst[s1&1]
    auto stage_async = [&](int s1) {
        int buf = s1 & 1;
        for (int cid = tid; cid < 256; cid += 128) {
            int r = cid >> 4, part = cid & 15;
            int b = b0 + r;
            int e = map[(size_t)b * VV + s1];
            const f16* src = (part < 8 && e >= 0)
                ? mbuf + (size_t)e * HH + (part & 7) * 8
                : xold + ((size_t)b * VV + s1) * HH + (part & 7) * 8;
            unsigned lds = (unsigned)(uintptr_t)&Xst[buf][r][part * 8];
            async_ld_b128(lds, src);
        }
    };

    stage_async(0);
    __syncthreads();

    for (int s = 0; s < VV; ++s) {
        if (s + 1 < VV) {
            stage_async(s + 1);                           // prefetch next step
            asm volatile("s_wait_asynccnt 0x2" ::: "memory"); // step-s pair done
        } else {
            asm volatile("s_wait_asynccnt 0x0" ::: "memory");
        }
        __syncthreads();

        const int buf = s & 1;
        v16h A[6];
#pragma unroll
        for (int c = 0; c < 4; ++c)
            A[c] = load_frag16(&Xst[buf][idx][c * 32 + sub * 8]);
        A[4] = load_frag16(&Hst[idx][sub * 8]);
        A[5] = load_frag16(&Hst[idx][32 + sub * 8]);

#pragma unroll
        for (int i = 0; i < 3; ++i) {
            int tn = t0 + i;
            if (tn < 8) {                       // r,z gates: K-fused 192
                v8f acc = {};
#pragma unroll
                for (int c = 0; c < 6; ++c) acc = wmma_f16(A[c], B[i][c], acc);
#pragma unroll
                for (int r = 0; r < 8; ++r) Grz[r + sub * 8][tn * 16 + idx] = acc[r];
            } else {                            // n gate: inn(K=128) + hn(K=64)
                v8f ai = {}, ah = {};
#pragma unroll
                for (int c = 0; c < 4; ++c) ai = wmma_f16(A[c], B[i][c], ai);
#pragma unroll
                for (int c = 4; c < 6; ++c) ah = wmma_f16(A[c], B[i][c], ah);
                int nb = (tn - 8) * 16 + idx;
#pragma unroll
                for (int r = 0; r < 8; ++r) {
                    Ginn[r + sub * 8][nb] = ai[r];
                    Ghn[r + sub * 8][nb] = ah[r];
                }
            }
        }
        __syncthreads();

        for (int f = tid; f < 1024; f += 128) {
            int br = f >> 6, o = f & 63;
            float rg = sigf(Grz[br][o] + b_ih[o] + b_hh[o]);
            float zg = sigf(Grz[br][64 + o] + b_ih[64 + o] + b_hh[64 + o]);
            float ng = tanhf(Ginn[br][o] + b_ih[128 + o] + rg * (Ghn[br][o] + b_hh[128 + o]));
            float hp = Hc[br][o];
            float hv = (1.0f - zg) * ng + zg * hp;
            Hc[br][o] = hv;
            Hst[br][o] = (f16)hv;
            xout[((size_t)(b0 + br) * VV + s) * HH + o] = (f16)hv;
        }
        __syncthreads();
    }
}

// ---------------- mean over (i,j) ----------------
__global__ void zero_kernel(float* p, int n) {
    int i = blockIdx.x * blockDim.x + threadIdx.x;
    if (i < n) p[i] = 0.0f;
}

__global__ __launch_bounds__(256) void mean_kernel(const f16* __restrict__ x,
                                                   float* __restrict__ gsum) {
    __shared__ float red[256];
    int i = blockIdx.x;
    int t = threadIdx.x, h = t & 63, jq = t >> 6;
    float s = 0.f;
    for (int j = jq; j < VV; j += 4) s += (float)x[((size_t)i * VV + j) * HH + h];
    red[t] = s;
    __syncthreads();
    if (t < 64) atomicAdd(&gsum[h], red[t] + red[t + 64] + red[t + 128] + red[t + 192]);
}

// ---------------- path bi-GRU readout (one WG per path) ----------------
__global__ __launch_bounds__(256) void path_kernel(
    const f16* __restrict__ x, const int* __restrict__ paths,
    const float* __restrict__ demands, const float* __restrict__ wd,
    const float* __restrict__ gsum,
    const float* __restrict__ fw_ih, const float* __restrict__ fw_hh,
    const float* __restrict__ fb_ih, const float* __restrict__ fb_hh,
    const float* __restrict__ bw_ih, const float* __restrict__ bw_hh,
    const float* __restrict__ bb_ih, const float* __restrict__ bb_hh,
    const float* __restrict__ wq, const float* __restrict__ bq,
    float* __restrict__ logits) {
    __shared__ float feat[7][DD];
    __shared__ float h[DD], hf[DD], ge[DD];
    __shared__ float gi[G3], gh[G3];
    int p = blockIdx.x, t = threadIdx.x;

    for (int q = t; q < 7 * DD; q += 256) {
        int s = q / DD, k = q % DD;
        float v;
        if (k < 64) {
            int src = paths[p * 8 + s], dst = paths[p * 8 + s + 1];
            v = (float)x[((size_t)src * VV + dst) * HH + k];
        } else {
            float de = demands[p] * wd[k - 64];
            v = de > 0.f ? de : 0.f;
        }
        feat[s][k] = v;
    }
    if (t < DD) {
        float g = (t < 64) ? gsum[t] * (1.0f / (float)(VV * VV)) : 0.0f;
        ge[t] = g;
        h[t] = g;
    }
    __syncthreads();

    // forward
    for (int s = 0; s < 7; ++s) {
        if (t < G3) {
            float a = fb_ih[t], b = fb_hh[t];
            for (int k = 0; k < DD; ++k) {
                a += feat[s][k] * fw_ih[t * DD + k];
                b += h[k] * fw_hh[t * DD + k];
            }
            gi[t] = a; gh[t] = b;
        }
        __syncthreads();
        float hn2 = 0.f;
        if (t < DD) {
            float r = sigf(gi[t] + gh[t]);
            float z = sigf(gi[DD + t] + gh[DD + t]);
            float n = tanhf(gi[2 * DD + t] + r * gh[2 * DD + t]);
            hn2 = (1.0f - z) * n + z * h[t];
        }
        __syncthreads();
        if (t < DD) h[t] = hn2;
        __syncthreads();
    }
    if (t < DD) { hf[t] = h[t]; h[t] = ge[t]; }
    __syncthreads();

    // backward (reversed sequence)
    for (int s = 6; s >= 0; --s) {
        if (t < G3) {
            float a = bb_ih[t], b = bb_hh[t];
            for (int k = 0; k < DD; ++k) {
                a += feat[s][k] * bw_ih[t * DD + k];
                b += h[k] * bw_hh[t * DD + k];
            }
            gi[t] = a; gh[t] = b;
        }
        __syncthreads();
        float hn2 = 0.f;
        if (t < DD) {
            float r = sigf(gi[t] + gh[t]);
            float z = sigf(gi[DD + t] + gh[DD + t]);
            float n = tanhf(gi[2 * DD + t] + r * gh[2 * DD + t]);
            hn2 = (1.0f - z) * n + z * h[t];
        }
        __syncthreads();
        if (t < DD) h[t] = hn2;
        __syncthreads();
    }

    if (t == 0) {
        float acc = bq[0];
        for (int k = 0; k < DD; ++k) acc += 0.5f * (hf[k] + h[k]) * wq[k];
        logits[p] = acc;
    }
}

__global__ void softmax_kernel(const float* __restrict__ logits, float* __restrict__ out) {
    __shared__ float red[64];
    int t = threadIdx.x;
    float l = logits[t];
    red[t] = l;
    __syncthreads();
    for (int off = 32; off >= 1; off >>= 1) {
        if (t < off) red[t] = fmaxf(red[t], red[t + off]);
        __syncthreads();
    }
    float m = red[0];
    __syncthreads();
    float e = expf(l - m);
    red[t] = e;
    __syncthreads();
    for (int off = 32; off >= 1; off >>= 1) {
        if (t < off) red[t] += red[t + off];
        __syncthreads();
    }
    out[t] = e / red[0];
}

extern "C" void kernel_launch(void* const* d_in, const int* in_sizes, int n_in,
                              void* d_out, int out_size, void* d_ws, size_t ws_size,
                              hipStream_t stream) {
    const float* adj     = (const float*)d_in[0];
    const int*   edges   = (const int*)d_in[1];
    const int*   paths   = (const int*)d_in[2];
    const float* demands = (const float*)d_in[3];
    const float* emb_w   = (const float*)d_in[4];
    const float* w1s     = (const float*)d_in[5];
    const float* w2s     = (const float*)d_in[6];
    const float* w3s     = (const float*)d_in[7];
    const float* g_wih   = (const float*)d_in[8];
    const float* g_whh   = (const float*)d_in[9];
    const float* g_bih   = (const float*)d_in[10];
    const float* g_bhh   = (const float*)d_in[11];
    const float* pf_wih  = (const float*)d_in[12];
    const float* pf_whh  = (const float*)d_in[13];
    const float* pf_bih  = (const float*)d_in[14];
    const float* pf_bhh  = (const float*)d_in[15];
    const float* pb_wih  = (const float*)d_in[16];
    const float* pb_whh  = (const float*)d_in[17];
    const float* pb_bih  = (const float*)d_in[18];
    const float* pb_bhh  = (const float*)d_in[19];
    const float* wq      = (const float*)d_in[20];
    const float* bq      = (const float*)d_in[21];
    const float* wd      = (const float*)d_in[22];
    float* out = (float*)d_out;

    char* ws = (char*)d_ws;
    const size_t XB = (size_t)VV * VV * HH * sizeof(f16);     // 32 MB
    size_t off = 0;
    f16* xb0 = (f16*)(ws + off); off += XB;
    f16* xb1 = (f16*)(ws + off); off += XB;
    f16* xb[2] = {xb0, xb1};
    f16*   mbuf    = (f16*)(ws + off);   off += (size_t)EE * HH * sizeof(f16);
    int*   map     = (int*)(ws + off);   off += (size_t)VV * VV * sizeof(int);
    f16*   feat    = (f16*)(ws + off);   off += (size_t)EE * 192 * sizeof(f16);
    float* col_sum = (float*)(ws + off); off += (size_t)VV * HH * sizeof(float);
    float* col_nz  = (float*)(ws + off); off += (size_t)VV * sizeof(float) + 256;
    f16*   wgru    = (f16*)(ws + off);   off += (size_t)192 * 192 * sizeof(f16);
    f16*   wmp     = (f16*)(ws + off);   off += (size_t)64 * 192 * sizeof(f16);
    float* gsum    = (float*)(ws + off); off += 256;
    float* logits  = (float*)(ws + off); off += 256;
    (void)ws_size; (void)in_sizes; (void)n_in; (void)out_size;

    embed_kernel<<<65536, 256, 0, stream>>>(adj, emb_w, xb[0]);
    conv_wgru_kernel<<<144, 256, 0, stream>>>(g_wih, g_whh, wgru);
    init_map_kernel<<<1024, 256, 0, stream>>>(map);
    build_map_kernel<<<EE / 256, 256, 0, stream>>>(edges, map);

    int icur = 0;
    for (int t = 0; t < 3; ++t) {
        conv_wmp_kernel<<<48, 256, 0, stream>>>(w1s + t * 4096, w2s + t * 4096,
                                                w3s + t * 4096, wmp);
        col_sum_kernel<<<VV, 256, 0, stream>>>(xb[icur], col_sum, col_nz);
        edge_feat_kernel<<<EE / 8, 256, 0, stream>>>(xb[icur], col_sum, col_nz,
                                                     edges, feat);
        mp_gemm_kernel<<<EE / 64, 128, 0, stream>>>(feat, wmp, mbuf);
        gru_kernel<<<VV / 16, 128, 0, stream>>>(mbuf, map, xb[icur], wgru,
                                                g_bih, g_bhh, xb[1 - icur]);
        icur = 1 - icur;
    }

    zero_kernel<<<1, 64, 0, stream>>>(gsum, 64);
    mean_kernel<<<VV, 256, 0, stream>>>(xb[icur], gsum);
    path_kernel<<<PP, 256, 0, stream>>>(xb[icur], paths, demands, wd, gsum,
                                        pf_wih, pf_whh, pf_bih, pf_bhh,
                                        pb_wih, pb_whh, pb_bih, pb_bhh,
                                        wq, bq, logits);
    softmax_kernel<<<1, 64, 0, stream>>>(logits, out);
}
